// MultiHeadAttention_17678085391071
// MI455X (gfx1250) — matrix operand order repaired and checked by reading, not compiled, not run on hardware
//
#include <hip/hip_runtime.h>

// MHA forward for MI455X (gfx1250), compute-bound (~412 GFLOP vs ~0.3 GB),
// so everything runs on v_wmma_f32_16x16x32_bf16 with f32 accumulation.
// GEMMs and attention stage tiles through LDS with CDNA5 async global->LDS
// copies (ASYNCcnt-tracked), double-buffered against the WMMA pipeline.
// All staged LDS rows are padded (+16B) to skew the 64-bank LDS.

#define D_MODEL 2048
#define SEQ_T   2048
#define BATCH   4
#define HEADS   16
#define HD      128
#define MROWS   (BATCH * SEQ_T)   // 8192

#define LDP_AB  72    // padded row (ushorts) for 64-wide tiles
#define LDP_K   136   // padded row (ushorts) for 128-wide K tiles

typedef __attribute__((ext_vector_type(16))) __bf16 v16bf;
typedef __attribute__((ext_vector_type(8)))  float  v8f;

union BFrag {
    v16bf v;
    uint4 q[2];
    unsigned short u[16];
};

__device__ __forceinline__ unsigned short f2bf(float f) {
    unsigned int u = __float_as_uint(f);
    u += 0x7fffu + ((u >> 16) & 1u);   // round-to-nearest-even
    return (unsigned short)(u >> 16);
}

__device__ __forceinline__ v8f wmma_bf16(v16bf a, v16bf b, v8f c) {
    return __builtin_amdgcn_wmma_f32_16x16x32_bf16(
        /*neg_a=*/false, a, /*neg_b=*/false, b,
        /*c_mod=*/(short)0, c, /*reuse_a=*/false, /*reuse_b=*/false);
}

// ---- CDNA5 async global->LDS copy (16B per lane), tracked by ASYNCcnt -----
__device__ __forceinline__ void async_cp16(unsigned int lds_off,
                                           const unsigned short* g) {
    asm volatile("global_load_async_to_lds_b128 %0, %1, off"
                 :: "v"(lds_off), "v"((unsigned long long)(uintptr_t)g)
                 : "memory");
}
__device__ __forceinline__ void wait_async0() {
    asm volatile("s_wait_asynccnt 0" ::: "memory");
}
// Low 32 bits of a generic LDS pointer are the LDS byte address.
__device__ __forceinline__ unsigned int lds_off32(const void* p) {
    return (unsigned int)(uintptr_t)p;
}

// A operand 16x32 bf16 from row-major [.., ld] matrix.
// lane l: m = l&15; k-chunks at hi*8 and hi*8+16 (ISA 16-bit A layout).
__device__ __forceinline__ v16bf load_a(const unsigned short* A, int ld,
                                        int row_base, int k0, int lane) {
    const int m = lane & 15, hi = lane >> 4;
    const unsigned short* p = A + (size_t)(row_base + m) * ld + k0 + hi * 8;
    BFrag f;
    f.q[0] = *reinterpret_cast<const uint4*>(p);
    f.q[1] = *reinterpret_cast<const uint4*>(p + 16);
    return f.v;
}

// B operand 32x16 bf16 from row-major Bt[N][K] (i.e. B transposed in memory).
// lane l: n = l&15; k-chunks at hi*8 and hi*8+16 (same order as A).
__device__ __forceinline__ v16bf load_b(const unsigned short* Bt, int ld,
                                        int n_base, int k0, int lane) {
    const int n = lane & 15, hi = lane >> 4;
    const unsigned short* p = Bt + (size_t)(n_base + n) * ld + k0 + hi * 8;
    BFrag f;
    f.q[0] = *reinterpret_cast<const uint4*>(p);
    f.q[1] = *reinterpret_cast<const uint4*>(p + 16);
    return f.v;
}

// A operand from this wave's 16x64 bf16 tile in LDS (padded ld = LDP_AB).
__device__ __forceinline__ v16bf load_pa(const unsigned short* sp, int lane, int k0) {
    const int m = lane & 15, hi = lane >> 4;
    const unsigned short* p = sp + m * LDP_AB + k0 + hi * 8;
    BFrag f;
    f.q[0] = *reinterpret_cast<const uint4*>(p);
    f.q[1] = *reinterpret_cast<const uint4*>(p + 16);
    return f.v;
}

// ---------------- Stage 1: f32 -> bf16 elementwise -------------------------
__global__ __launch_bounds__(256) void cvt_bf16_kernel(
        const float* __restrict__ x, unsigned short* __restrict__ y, int n) {
    int i = (blockIdx.x * 256 + threadIdx.x) * 4;
    if (i + 3 < n) {
        float4 f = *reinterpret_cast<const float4*>(x + i);
        y[i + 0] = f2bf(f.x);
        y[i + 1] = f2bf(f.y);
        y[i + 2] = f2bf(f.z);
        y[i + 3] = f2bf(f.w);
    }
}

// ---------------- Stage 2: W[K][N] f32 -> Wt[N][K] bf16 (LDS tiled) --------
__global__ __launch_bounds__(256) void transpose_w_kernel(
        const float* __restrict__ W, unsigned short* __restrict__ Wt) {
    __shared__ float tile[64][65];
    const int bx = blockIdx.x * 64;  // n
    const int by = blockIdx.y * 64;  // k
    const int tx = threadIdx.x & 63, ty = threadIdx.x >> 6;
#pragma unroll
    for (int i = 0; i < 64; i += 4)
        tile[ty + i][tx] = W[(size_t)(by + ty + i) * D_MODEL + bx + tx];
    __syncthreads();
#pragma unroll
    for (int i = 0; i < 64; i += 4)
        Wt[(size_t)(bx + ty + i) * D_MODEL + by + tx] = f2bf(tile[tx][ty + i]);
}

// ---------------- Stage 3/5: WMMA GEMM 8192x2048x2048 ----------------------
// A: bf16 [8192][2048] row-major.  Bt: bf16 [2048][2048] = B transposed.
// 128x128 block tile, K staged 64-deep through double-buffered LDS with
// async global->LDS copies overlapped against WMMA on the other buffer.
// MODE 0: out bf16 Q/K layout  [b,h][t][hd]
// MODE 1: out bf16 V  layout   [b,h][hd][t]   (pre-transposed for P@V)
// MODE 2: out f32 row-major + bias
template <int MODE>
__global__ __launch_bounds__(256) void gemm_bf16_kernel(
        const unsigned short* __restrict__ A, const unsigned short* __restrict__ Bt,
        void* __restrict__ Cout, const float* __restrict__ bias) {
    __shared__ unsigned short ldsA[2][128 * LDP_AB];   // ~18 KB per buffer
    __shared__ unsigned short ldsB[2][128 * LDP_AB];

    const int tid = threadIdx.x, lane = tid & 31, wid = tid >> 5;
    const int wrow = wid >> 1, wcol = wid & 1;            // 4x2 wave grid
    const int mblk = blockIdx.y * 128;
    const int nblk = blockIdx.x * 128;
    const int m15 = lane & 15, hi = lane >> 4;

    v8f c[2][4];
#pragma unroll
    for (int mt = 0; mt < 2; ++mt)
#pragma unroll
        for (int nt = 0; nt < 4; ++nt) c[mt][nt] = (v8f){};

    // stage one 128x64 A tile and one 128x64 B tile (1024 x 16B chunks,
    // 4 chunks per thread per tile) into LDS buffer `buf`
    auto stage = [&](int buf, int kt) {
#pragma unroll
        for (int i = 0; i < 4; ++i) {
            const int ch = tid + i * 256;
            const int r  = ch >> 3;
            const int ko = (ch & 7) * 8;
            async_cp16(lds_off32(&ldsA[buf][r * LDP_AB + ko]),
                       A + (size_t)(mblk + r) * D_MODEL + kt + ko);
            async_cp16(lds_off32(&ldsB[buf][r * LDP_AB + ko]),
                       Bt + (size_t)(nblk + r) * D_MODEL + kt + ko);
        }
    };

    stage(0, 0);
    wait_async0();
    __syncthreads();

    int buf = 0;
    for (int kt = 0; kt < D_MODEL; kt += 64) {
        if (kt + 64 < D_MODEL) stage(buf ^ 1, kt + 64);   // prefetch next tile

#pragma unroll
        for (int kc = 0; kc < 2; ++kc) {
            const int k0 = kc * 32;
            v16bf a0 = load_a(&ldsA[buf][0], LDP_AB, wrow * 32,      k0, lane);
            v16bf a1 = load_a(&ldsA[buf][0], LDP_AB, wrow * 32 + 16, k0, lane);
#pragma unroll
            for (int nt = 0; nt < 4; ++nt) {
                v16bf bf = load_b(&ldsB[buf][0], LDP_AB, wcol * 64 + nt * 16, k0, lane);
                c[0][nt] = wmma_bf16(a0, bf, c[0][nt]);
                c[1][nt] = wmma_bf16(a1, bf, c[1][nt]);
            }
        }

        wait_async0();        // next tile landed in LDS
        __syncthreads();      // all waves done reading current buffer
        buf ^= 1;
    }

#pragma unroll
    for (int mt = 0; mt < 2; ++mt)
#pragma unroll
        for (int nt = 0; nt < 4; ++nt)
#pragma unroll
            for (int r = 0; r < 8; ++r) {
                const int m = mblk + wrow * 32 + mt * 16 + r + 8 * hi;
                const int n = nblk + wcol * 64 + nt * 16 + m15;
                const float v = c[mt][nt][r];
                if (MODE == 2) {
                    reinterpret_cast<float*>(Cout)[(size_t)m * D_MODEL + n] =
                        v + bias[n];
                } else {
                    const int b = m >> 11, t = m & 2047;
                    const int h = n >> 7,  dd = n & 127;
                    unsigned short* o = reinterpret_cast<unsigned short*>(Cout);
                    if (MODE == 0)
                        o[(((size_t)(b * HEADS + h) * SEQ_T) + t) * HD + dd] = f2bf(v);
                    else
                        o[(((size_t)(b * HEADS + h) * HD) + dd) * SEQ_T + t] = f2bf(v);
                }
            }
}

// ---------------- Stage 4: flash attention (non-causal, per ref quirk) -----
// Q,K: bf16 [b,h][T][HD]; V: bf16 [b,h][HD][T].
// K/V key-block tiles are async-staged into double-buffered LDS and shared
// by all 8 waves (8x less global traffic than per-wave fetches).
// Output Mmix bf16: M[b][h*128+d][q]  (reproduces transpose(0,1,3,2).reshape).
__global__ __launch_bounds__(256) void attn_kernel(
        const unsigned short* __restrict__ Qb, const unsigned short* __restrict__ Kb,
        const unsigned short* __restrict__ Vt, unsigned short* __restrict__ Mmix) {
    const int tid = threadIdx.x, lane = tid & 31, wid = tid >> 5;
    const int bh = blockIdx.y;                 // 0..63
    const int b = bh >> 4, h = bh & 15;
    const unsigned short* Q = Qb + (size_t)bh * SEQ_T * HD;
    const unsigned short* K = Kb + (size_t)bh * SEQ_T * HD;
    const unsigned short* V = Vt + (size_t)bh * HD * SEQ_T;

    __shared__ unsigned short ldsK[2][64 * LDP_K];    // 64 keys x 128 dims
    __shared__ unsigned short ldsV[2][128 * LDP_AB];  // 128 dims x 64 keys
    __shared__ unsigned short Pld[8 * 16 * LDP_AB];   // per-wave 16x64 P tile
    unsigned short* myP = Pld + wid * (16 * LDP_AB);

    const int qbase = blockIdx.x * 128 + wid * 16;
    const int m15 = lane & 15, hi = lane >> 4;
    const float scale = 0.08838834764831845f;       // 1/sqrt(128)

    // async-stage one K tile (64x128) and one V tile (128x64) for key block kb2
    auto stageKV = [&](int buf, int kb2) {
#pragma unroll
        for (int i = 0; i < 4; ++i) {
            const int ch = tid + i * 256;
            const int rk = ch >> 4, kok = (ch & 15) * 8;   // K: 16 chunks/row
            async_cp16(lds_off32(&ldsK[buf][rk * LDP_K + kok]),
                       K + (size_t)(kb2 + rk) * HD + kok);
            const int rv = ch >> 3, kov = (ch & 7) * 8;    // V: 8 chunks/row
            async_cp16(lds_off32(&ldsV[buf][rv * LDP_AB + kov]),
                       V + (size_t)rv * SEQ_T + kb2 + kov);
        }
    };

    // Q fragments for this wave's 16 rows (hd=128 -> 4 K-chunks)
    v16bf qf[4];
#pragma unroll
    for (int kk = 0; kk < 4; ++kk) qf[kk] = load_a(Q, HD, qbase, kk * 32, lane);

    v8f O[8];
#pragma unroll
    for (int i = 0; i < 8; ++i) O[i] = (v8f){};
    float rowm[8], rowl[8];
#pragma unroll
    for (int r = 0; r < 8; ++r) { rowm[r] = -1e30f; rowl[r] = 0.f; }

    stageKV(0, 0);
    wait_async0();
    __syncthreads();

    int buf = 0;
    for (int kb = 0; kb < SEQ_T; kb += 64) {
        if (kb + 64 < SEQ_T) stageKV(buf ^ 1, kb + 64);   // prefetch next tiles

        // S = scale * Q @ K^T   (16 x 64) from LDS K tile
        v8f s[4];
#pragma unroll
        for (int nt = 0; nt < 4; ++nt) {
            v8f acc = (v8f){};
#pragma unroll
            for (int kk = 0; kk < 4; ++kk) {
                v16bf bf = load_b(&ldsK[buf][0], LDP_K, nt * 16, kk * 32, lane);
                acc = wmma_bf16(qf[kk], bf, acc);
            }
#pragma unroll
            for (int r = 0; r < 8; ++r) acc[r] *= scale;
            s[nt] = acc;
        }

        // online-softmax row max (reduce across the 16 lanes holding a row)
        float alpha[8];
#pragma unroll
        for (int r = 0; r < 8; ++r) {
            float v = fmaxf(fmaxf(s[0][r], s[1][r]), fmaxf(s[2][r], s[3][r]));
            v = fmaxf(v, __shfl_xor(v, 1, 32));
            v = fmaxf(v, __shfl_xor(v, 2, 32));
            v = fmaxf(v, __shfl_xor(v, 4, 32));
            v = fmaxf(v, __shfl_xor(v, 8, 32));
            const float nm = fmaxf(rowm[r], v);
            alpha[r] = __expf(rowm[r] - nm);
            rowm[r] = nm;
        }

        // P = exp(S - m), row sums, stage P in LDS (C-layout -> A-layout)
        float rs[8];
#pragma unroll
        for (int r = 0; r < 8; ++r) rs[r] = 0.f;
#pragma unroll
        for (int nt = 0; nt < 4; ++nt)
#pragma unroll
            for (int r = 0; r < 8; ++r) {
                const float p = __expf(s[nt][r] - rowm[r]);
                rs[r] += p;
                myP[(r + 8 * hi) * LDP_AB + nt * 16 + m15] = f2bf(p);
            }
#pragma unroll
        for (int r = 0; r < 8; ++r) {
            float v = rs[r];
            v += __shfl_xor(v, 1, 32);
            v += __shfl_xor(v, 2, 32);
            v += __shfl_xor(v, 4, 32);
            v += __shfl_xor(v, 8, 32);
            rowl[r] = rowl[r] * alpha[r] + v;
        }

        // rescale accumulators
#pragma unroll
        for (int nt2 = 0; nt2 < 8; ++nt2)
#pragma unroll
            for (int r = 0; r < 8; ++r) O[nt2][r] *= alpha[r];

        // O += P @ V from LDS V tile ([d][t] is exactly the B operand)
        v16bf pa0 = load_pa(myP, lane, 0);
        v16bf pa1 = load_pa(myP, lane, 32);
#pragma unroll
        for (int nt2 = 0; nt2 < 8; ++nt2) {
            v16bf vb0 = load_b(&ldsV[buf][0], LDP_AB, nt2 * 16, 0,  lane);
            v16bf vb1 = load_b(&ldsV[buf][0], LDP_AB, nt2 * 16, 32, lane);
            O[nt2] = wmma_bf16(pa0, vb0, O[nt2]);
            O[nt2] = wmma_bf16(pa1, vb1, O[nt2]);
        }

        wait_async0();        // next K/V tiles landed in LDS
        __syncthreads();      // all waves done reading current buffer
        buf ^= 1;
    }

    // finalize and store in the reference's mixed layout: M[b][h*128+d][q]
    float inv[8];
#pragma unroll
    for (int r = 0; r < 8; ++r) inv[r] = 1.f / rowl[r];
    const size_t ob = (size_t)b * D_MODEL * SEQ_T;
#pragma unroll
    for (int nt2 = 0; nt2 < 8; ++nt2) {
        const int d = nt2 * 16 + m15;
#pragma unroll
        for (int r = 0; r < 8; ++r) {
            const int q = qbase + r + 8 * hi;
            Mmix[ob + (size_t)(h * HD + d) * SEQ_T + q] = f2bf(O[nt2][r] * inv[r]);
        }
    }
}

// ---------------------------------------------------------------------------
extern "C" void kernel_launch(void* const* d_in, const int* in_sizes, int n_in,
                              void* d_out, int out_size, void* d_ws, size_t ws_size,
                              hipStream_t stream) {
    (void)in_sizes; (void)n_in; (void)out_size; (void)ws_size;
    const float* x  = (const float*)d_in[0];
    const float* Wq = (const float*)d_in[1];
    const float* Wk = (const float*)d_in[2];
    const float* Wv = (const float*)d_in[3];
    const float* Wo = (const float*)d_in[4];
    const float* bo = (const float*)d_in[5];

    char* ws = (char*)d_ws;
    const size_t SZ_X = (size_t)MROWS * D_MODEL * 2;          // 33.5 MB bf16
    const size_t SZ_W = (size_t)D_MODEL * D_MODEL * 2;        //  8.4 MB bf16
    unsigned short* xb   = (unsigned short*)(ws);
    unsigned short* Wqt  = (unsigned short*)(ws + SZ_X);
    unsigned short* Wkt  = (unsigned short*)(ws + SZ_X + SZ_W);
    unsigned short* Wvt  = (unsigned short*)(ws + SZ_X + 2 * SZ_W);
    unsigned short* Wot  = (unsigned short*)(ws + SZ_X + 3 * SZ_W);
    unsigned short* Qb   = (unsigned short*)(ws + SZ_X + 4 * SZ_W);
    unsigned short* Kb   = (unsigned short*)(ws + 2 * SZ_X + 4 * SZ_W);
    unsigned short* Vt   = (unsigned short*)(ws + 3 * SZ_X + 4 * SZ_W);
    unsigned short* Mmix = (unsigned short*)(ws + 4 * SZ_X + 4 * SZ_W);

    // 1) x -> bf16
    const int nx = MROWS * D_MODEL;
    cvt_bf16_kernel<<<nx / (4 * 256), 256, 0, stream>>>(x, xb, nx);

    // 2) weights -> bf16 transposed [out][in]
    dim3 tg(32, 32);
    transpose_w_kernel<<<tg, 256, 0, stream>>>(Wq, Wqt);
    transpose_w_kernel<<<tg, 256, 0, stream>>>(Wk, Wkt);
    transpose_w_kernel<<<tg, 256, 0, stream>>>(Wv, Wvt);
    transpose_w_kernel<<<tg, 256, 0, stream>>>(Wo, Wot);

    // 3) QKV projections (WMMA + async-LDS staging)
    dim3 gg(D_MODEL / 128, MROWS / 128);   // 16 x 64
    gemm_bf16_kernel<0><<<gg, 256, 0, stream>>>(xb, Wqt, Qb, nullptr);
    gemm_bf16_kernel<0><<<gg, 256, 0, stream>>>(xb, Wkt, Kb, nullptr);
    gemm_bf16_kernel<1><<<gg, 256, 0, stream>>>(xb, Wvt, Vt, nullptr);

    // 4) flash attention -> Mmix (reference's transposed-reshape layout)
    dim3 ga(SEQ_T / 128, BATCH * HEADS);   // 16 x 64
    attn_kernel<<<ga, 256, 0, stream>>>(Qb, Kb, Vt, Mmix);

    // 5) output projection + bias (f32 out)
    gemm_bf16_kernel<2><<<gg, 256, 0, stream>>>(Mmix, Wot, d_out, bo);
}
